// MultiHeadAttention_41025527612122
// MI455X (gfx1250) — compile-verified, matrix operand order checked by reading, not run on hardware
//
#include <hip/hip_runtime.h>

// ---------------------------------------------------------------------------
// Types for CDNA5 WMMA
// ---------------------------------------------------------------------------
typedef __attribute__((ext_vector_type(16))) __bf16       v16bf;
typedef __attribute__((ext_vector_type(8)))  float        v8f;
typedef __attribute__((ext_vector_type(4)))  unsigned int u32x4;

union Frag16 {
    u32x4 q[2];   // two 16-byte halves (K pairs 0..7, K pairs 16..23)
    v16bf v;      // operand for v_wmma_f32_16x16x32_bf16
};

static __device__ __forceinline__ unsigned short f2bf(float f) {
    unsigned int u = __float_as_uint(f);
    if ((u & 0x7fffffffu) > 0x7f800000u) return 0x7fc0u;          // NaN
    unsigned int r = u + 0x7fffu + ((u >> 16) & 1u);              // RNE
    return (unsigned short)(r >> 16);
}
static __device__ __forceinline__ float bf2f(unsigned short h) {
    return __uint_as_float(((unsigned int)h) << 16);
}

// ---------------------------------------------------------------------------
// CDNA5 async global->LDS copy (ASYNCcnt) via inline asm; sync fallback for
// non-AMDGCN (host) pass.
//   global_load_async_to_lds_b128: VDST = per-lane LDS byte offset VGPR,
//   VADDR = 64-bit global address (GV mode).
//   Low 32 bits of a generic pointer to __shared__ are the AS3 offset.
// ---------------------------------------------------------------------------
#if defined(__AMDGCN__)
#define HAVE_ASYNC_LDS 1
static __device__ __forceinline__ void async_cp16(const unsigned short* g, unsigned short* l) {
    unsigned loff = (unsigned)(unsigned long long)l;
    asm volatile("global_load_async_to_lds_b128 %0, %1, off"
                 :: "v"(loff), "v"(g)
                 : "memory");
}
static __device__ __forceinline__ void wait_async0() {
    asm volatile("s_wait_asynccnt 0x0" ::: "memory");
}
#endif

// ---------------------------------------------------------------------------
// fp32 -> bf16 conversion, 4 elements / thread (row-major preserved)
// ---------------------------------------------------------------------------
__global__ __launch_bounds__(256) void cvt_f32_to_bf16(
    const float* __restrict__ in, unsigned short* __restrict__ out, long n4)
{
    long i = (long)blockIdx.x * 256 + threadIdx.x;
    if (i >= n4) return;
    float4 v = ((const float4*)in)[i];
    union { unsigned long long u; unsigned short s[4]; } p;
    p.s[0] = f2bf(v.x); p.s[1] = f2bf(v.y); p.s[2] = f2bf(v.z); p.s[3] = f2bf(v.w);
    ((unsigned long long*)out)[i] = p.u;
}

// ---------------------------------------------------------------------------
// fp32 [K][N] -> bf16 transposed [N][K]   (32x32 tiles via LDS)
// ---------------------------------------------------------------------------
__global__ __launch_bounds__(256) void cvt_transpose_f32_to_bf16(
    const float* __restrict__ W, unsigned short* __restrict__ Wt, int K, int N)
{
    __shared__ __attribute__((aligned(16))) unsigned short t[32 * 40];
    const int tid = threadIdx.x;
    const int k0 = blockIdx.x * 32, n0 = blockIdx.y * 32;

    {   // read 32x32 fp32 tile, convert, store transposed into LDS
        int k = tid >> 3, nc = (tid & 7) * 4;
        float4 v = *(const float4*)(W + (size_t)(k0 + k) * N + n0 + nc);
        t[(nc + 0) * 40 + k] = f2bf(v.x);
        t[(nc + 1) * 40 + k] = f2bf(v.y);
        t[(nc + 2) * 40 + k] = f2bf(v.z);
        t[(nc + 3) * 40 + k] = f2bf(v.w);
    }
    __syncthreads();
    {   // write [N][K] tile with 8B stores
        int n = tid >> 3, kc = (tid & 7) * 4;
        union { unsigned long long u; unsigned short s[4]; } p;
        p.s[0] = t[n * 40 + kc + 0];
        p.s[1] = t[n * 40 + kc + 1];
        p.s[2] = t[n * 40 + kc + 2];
        p.s[3] = t[n * 40 + kc + 3];
        *(unsigned long long*)(Wt + (size_t)(n0 + n) * K + k0 + kc) = p.u;
    }
}

// ---------------------------------------------------------------------------
// bf16 GEMM, fp32 accumulate:  C[M,N] = A[M,K] @ Bt[N,K]^T + bias
//   A row-major bf16 [M][K]; Bt row-major bf16 [N][K] (pre-transposed).
// Block: 256 thr = 8 waves; block tile 128x128x32; wave tile 32x64 (8 WMMA).
// Double-buffered LDS; async global->LDS staging.
// ---------------------------------------------------------------------------
#define BM 128
#define BN 128
#define BK 32
#define LSTR 40   // LDS row stride in shorts: 80B, 16B-aligned, conflict-free

__global__ __launch_bounds__(256) void gemm_bf16_wmma(
    const unsigned short* __restrict__ A,
    const unsigned short* __restrict__ Bt,
    const float*          __restrict__ bias,
    float*                __restrict__ Cf,
    unsigned short*       __restrict__ Cb,
    int M, int N, int K)
{
    __shared__ __attribute__((aligned(16))) unsigned short lA[2][BM * LSTR];
    __shared__ __attribute__((aligned(16))) unsigned short lB[2][BN * LSTR];

    const int tid   = threadIdx.x;
    const int lane  = tid & 31;
    const int wid   = tid >> 5;
    const int waveM = wid & 3;                   // 0..3 -> 32-row slice
    const int waveN = wid >> 2;                  // 0..1 -> 64-col slice
    const long m0 = (long)blockIdx.x * BM;
    const long n0 = (long)blockIdx.y * BN;

    v8f acc[2][4] = {};

    // staging coords: each thread copies one 16-short half-row of A and Bt
    const int srow = tid >> 1;                   // 0..127
    const int sseg = (tid & 1) * 16;             // 0 or 16 shorts
    const unsigned short* gA = A  + (m0 + srow) * (long)K + sseg;
    const unsigned short* gB = Bt + (n0 + srow) * (long)K + sseg;

    // fragment addressing (ISA 16-bit operand layout, wave32)
    const int fr = lane & 15;
    const int kb = (lane >> 4) << 3;             // 0 or 8

    const int NK = K / BK;

#if defined(HAVE_ASYNC_LDS)
    // prologue: async stage tile 0 into buffer 0
    {
        async_cp16(gA,     &lA[0][srow * LSTR + sseg]);
        async_cp16(gA + 8, &lA[0][srow * LSTR + sseg + 8]);
        async_cp16(gB,     &lB[0][srow * LSTR + sseg]);
        async_cp16(gB + 8, &lB[0][srow * LSTR + sseg + 8]);
    }
    for (int kt = 0; kt < NK; ++kt) {
        wait_async0();        // own async writes for tile kt landed
        __syncthreads();      // DScnt-wait + barrier: all reads of other buf done
        if (kt + 1 < NK) {
            const int nb = (kt + 1) & 1;
            const int ko = (kt + 1) * BK;
            async_cp16(gA + ko,     &lA[nb][srow * LSTR + sseg]);
            async_cp16(gA + ko + 8, &lA[nb][srow * LSTR + sseg + 8]);
            async_cp16(gB + ko,     &lB[nb][srow * LSTR + sseg]);
            async_cp16(gB + ko + 8, &lB[nb][srow * LSTR + sseg + 8]);
        }
        const int cb = kt & 1;
        Frag16 fa[2], fb[4];
        #pragma unroll
        for (int i = 0; i < 2; ++i) {
            int r = waveM * 32 + i * 16 + fr;
            fa[i].q[0] = *(const u32x4*)&lA[cb][r * LSTR + kb];
            fa[i].q[1] = *(const u32x4*)&lA[cb][r * LSTR + kb + 16];
        }
        #pragma unroll
        for (int j = 0; j < 4; ++j) {
            int c = waveN * 64 + j * 16 + fr;
            fb[j].q[0] = *(const u32x4*)&lB[cb][c * LSTR + kb];
            fb[j].q[1] = *(const u32x4*)&lB[cb][c * LSTR + kb + 16];
        }
        #pragma unroll
        for (int i = 0; i < 2; ++i)
            #pragma unroll
            for (int j = 0; j < 4; ++j)
                acc[i][j] = __builtin_amdgcn_wmma_f32_16x16x32_bf16(
                    false, fa[i].v, false, fb[j].v,
                    (short)0, acc[i][j], false, false);
    }
#else
    // sync fallback (host pass / no async support)
    {
        u32x4 a0 = *(const u32x4*)gA, a1 = *(const u32x4*)(gA + 8);
        u32x4 b0 = *(const u32x4*)gB, b1 = *(const u32x4*)(gB + 8);
        *(u32x4*)&lA[0][srow * LSTR + sseg]     = a0;
        *(u32x4*)&lA[0][srow * LSTR + sseg + 8] = a1;
        *(u32x4*)&lB[0][srow * LSTR + sseg]     = b0;
        *(u32x4*)&lB[0][srow * LSTR + sseg + 8] = b1;
    }
    for (int kt = 0; kt < NK; ++kt) {
        __syncthreads();
        const int cb = kt & 1;
        Frag16 fa[2], fb[4];
        #pragma unroll
        for (int i = 0; i < 2; ++i) {
            int r = waveM * 32 + i * 16 + fr;
            fa[i].q[0] = *(const u32x4*)&lA[cb][r * LSTR + kb];
            fa[i].q[1] = *(const u32x4*)&lA[cb][r * LSTR + kb + 16];
        }
        #pragma unroll
        for (int j = 0; j < 4; ++j) {
            int c = waveN * 64 + j * 16 + fr;
            fb[j].q[0] = *(const u32x4*)&lB[cb][c * LSTR + kb];
            fb[j].q[1] = *(const u32x4*)&lB[cb][c * LSTR + kb + 16];
        }
        #pragma unroll
        for (int i = 0; i < 2; ++i)
            #pragma unroll
            for (int j = 0; j < 4; ++j)
                acc[i][j] = __builtin_amdgcn_wmma_f32_16x16x32_bf16(
                    false, fa[i].v, false, fb[j].v,
                    (short)0, acc[i][j], false, false);
        if (kt + 1 < NK) {
            const int nb = (kt + 1) & 1;
            const int ko = (kt + 1) * BK;
            u32x4 a0 = *(const u32x4*)(gA + ko), a1 = *(const u32x4*)(gA + ko + 8);
            u32x4 b0 = *(const u32x4*)(gB + ko), b1 = *(const u32x4*)(gB + ko + 8);
            *(u32x4*)&lA[nb][srow * LSTR + sseg]     = a0;
            *(u32x4*)&lA[nb][srow * LSTR + sseg + 8] = a1;
            *(u32x4*)&lB[nb][srow * LSTR + sseg]     = b0;
            *(u32x4*)&lB[nb][srow * LSTR + sseg + 8] = b1;
        }
    }
#endif

    // ---- epilogue: C/D layout (VGPR r -> M = r + 8*(lane>=16), N = lane&15)
    const int fn = lane & 15;
    const int mo = (lane >> 4) << 3;
    #pragma unroll
    for (int i = 0; i < 2; ++i) {
        #pragma unroll
        for (int j = 0; j < 4; ++j) {
            long n = n0 + waveN * 64 + j * 16 + fn;
            float bv = bias ? bias[n] : 0.0f;
            #pragma unroll
            for (int r = 0; r < 8; ++r) {
                long m = m0 + waveM * 32 + i * 16 + mo + r;
                float val = acc[i][j][r] + bv;
                if (Cf) Cf[m * (long)N + n] = val;
                else    Cb[m * (long)N + n] = f2bf(val);
            }
        }
    }
}

// ---------------------------------------------------------------------------
// Per-token attention over heads (negligible FLOPs; VALU fp32).
//   qkv[T,3072] bf16: head h -> q=[h*192,+64), k=+64, v=+128
// 1 wave per token, 8 tokens / block. LDS: 48KB qkv + 8KB attn.
// ---------------------------------------------------------------------------
__global__ __launch_bounds__(256) void attn_heads_kernel(
    const unsigned short* __restrict__ qkv,
    unsigned short*       __restrict__ out,
    int T)
{
    __shared__ __attribute__((aligned(16))) unsigned short s_qkv[8][3072];
    __shared__ float s_att[8][256];

    const int tid  = threadIdx.x;
    const int lane = tid & 31;
    const int wid  = tid >> 5;
    const int t    = blockIdx.x * 8 + wid;

    {
        const u32x4* g = (const u32x4*)(qkv + (long)t * 3072);
        u32x4* s = (u32x4*)s_qkv[wid];
        for (int i = lane; i < 384; i += 32) s[i] = g[i];
    }
    __syncthreads();

    const int h     = lane >> 1;
    const int gbase = (lane & 1) * 8;
    {
        float sc[8] = {};
        for (int d = 0; d < 64; ++d) {
            float qv = bf2f(s_qkv[wid][h * 192 + d]);
            #pragma unroll
            for (int j = 0; j < 8; ++j)
                sc[j] += qv * bf2f(s_qkv[wid][(gbase + j) * 192 + 64 + d]);
        }
        #pragma unroll
        for (int j = 0; j < 8; ++j)
            s_att[wid][h * 16 + gbase + j] = sc[j] * 0.125f;   // 1/sqrt(64)
    }
    __syncthreads();

    if (lane < 16) {
        float* rowp = &s_att[wid][lane * 16];
        float mx = rowp[0];
        #pragma unroll
        for (int g = 1; g < 16; ++g) mx = fmaxf(mx, rowp[g]);
        float e[16], sum = 0.0f;
        #pragma unroll
        for (int g = 0; g < 16; ++g) { e[g] = __expf(rowp[g] - mx); sum += e[g]; }
        float inv = 1.0f / sum;
        #pragma unroll
        for (int g = 0; g < 16; ++g) rowp[g] = e[g] * inv;
    }
    __syncthreads();

    unsigned short* orow = out + (long)t * 1024;
    for (int i = 0; i < 32; ++i) {
        int o  = i * 32 + lane;
        int oh = o >> 6, od = o & 63;
        float a = 0.0f;
        #pragma unroll
        for (int g = 0; g < 16; ++g)
            a += s_att[wid][oh * 16 + g] * bf2f(s_qkv[wid][g * 192 + 128 + od]);
        orow[o] = f2bf(a);
    }
}

// ---------------------------------------------------------------------------
// Launch
// ---------------------------------------------------------------------------
extern "C" void kernel_launch(void* const* d_in, const int* in_sizes, int n_in,
                              void* d_out, int out_size, void* d_ws, size_t ws_size,
                              hipStream_t stream) {
    (void)in_sizes; (void)n_in; (void)out_size; (void)ws_size;

    const float* x      = (const float*)d_in[0];   // [32768,1024]
    const float* w_qkv  = (const float*)d_in[1];   // [1024,3072]
    const float* b_qkv  = (const float*)d_in[2];   // [3072]
    const float* w_out  = (const float*)d_in[3];   // [1024,1024]
    const float* b_out  = (const float*)d_in[4];   // [1024]
    float* out          = (float*)d_out;           // [32768,1024]

    const int M  = 4 * 8192;   // 32768 tokens
    const int E  = 1024;
    const int N1 = 3072;

    // workspace (bf16 = unsigned short), all segments 16B-aligned
    unsigned short* xb     = (unsigned short*)d_ws;          // [M][E]
    unsigned short* wqkvT  = xb    + (size_t)M * E;          // [N1][E]
    unsigned short* woutT  = wqkvT + (size_t)E * N1;         // [E][E]
    unsigned short* qkvb   = woutT + (size_t)E * E;          // [M][N1]
    unsigned short* aob    = qkvb  + (size_t)M * N1;         // [M][E]

    // 1) conversions (x flat; weights converted AND transposed to [N][K])
    {
        long n4 = (long)M * E / 4;
        cvt_f32_to_bf16<<<(n4 + 255) / 256, 256, 0, stream>>>(x, xb, n4);
        cvt_transpose_f32_to_bf16<<<dim3(E / 32, N1 / 32), 256, 0, stream>>>(
            w_qkv, wqkvT, E, N1);
        cvt_transpose_f32_to_bf16<<<dim3(E / 32, E / 32), 256, 0, stream>>>(
            w_out, woutT, E, E);
    }

    // 2) QKV GEMM: qkv = x @ w_qkv + b_qkv  (bf16 out)
    gemm_bf16_wmma<<<dim3(M / BM, N1 / BN), 256, 0, stream>>>(
        xb, wqkvT, b_qkv, nullptr, qkvb, M, N1, E);

    // 3) per-token attention over heads (bf16 out)
    attn_heads_kernel<<<M / 8, 256, 0, stream>>>(qkvb, aob, M);

    // 4) output GEMM: out = attn_out @ w_out + b_out  (fp32 out)
    gemm_bf16_wmma<<<dim3(M / BM, E / BN), 256, 0, stream>>>(
        aob, woutT, b_out, out, nullptr, M, E, E);
}
